// DualAttentionEncoderBlock_51135880626863
// MI455X (gfx1250) — matrix operand order repaired and checked by reading, not compiled, not run on hardware
//
#include <hip/hip_runtime.h>

// ---------------------------------------------------------------------------
// DualAttentionEncoderBlock for MI455X (gfx1250).
// bf16 WMMA + Tensor Data Mover (TDM) tile staging, double-buffered LDS.
// B=4, L=1024, D=768, H=16, hd_seq=48, hd_chan=64, MAX_DIST=1024.
// ---------------------------------------------------------------------------

typedef __bf16   bf16x16 __attribute__((ext_vector_type(16)));
typedef __bf16   bf16x8  __attribute__((ext_vector_type(8)));
typedef float    f32x8   __attribute__((ext_vector_type(8)));
typedef unsigned u32x4   __attribute__((ext_vector_type(4)));
typedef unsigned u32x8   __attribute__((ext_vector_type(8)));

__device__ __forceinline__ __bf16 f2bf(float f) {
    unsigned u = __float_as_uint(f);
    u += 0x7FFFu + ((u >> 16) & 1u);          // round-to-nearest-even
    return __builtin_bit_cast(__bf16, (unsigned short)(u >> 16));
}
__device__ __forceinline__ float bf2f(__bf16 b) {
    unsigned short s = __builtin_bit_cast(unsigned short, b);
    return __uint_as_float((unsigned)s << 16);
}
__device__ __forceinline__ __bf16 bf_zero() {
    return __builtin_bit_cast(__bf16, (unsigned short)0);
}

__device__ __forceinline__ void wait_tensorcnt0() {
#if __has_builtin(__builtin_amdgcn_s_wait_tensorcnt)
    __builtin_amdgcn_s_wait_tensorcnt(0);
#else
    asm volatile("s_wait_tensorcnt 0x0" ::: "memory");
#endif
}

// ---------------------------------------------------------------------------
// TDM: async-copy a 2D bf16 tile (tile_d1 rows x tile_d0 elems) from global
// (row stride `stride0` elements) into LDS at byte offset `lds_off`, with
// 16B padding appended after every 64B row (pad_interval=3 -> 16 DWORDs,
// pad_amount=3 -> 4 DWORDs) so LDS rows land on an 80B stride (LDK=40).
// Rows/cols beyond tensor_d1/tensor_d0 are zero-filled by TDM OOB handling.
// D# encoding per CDNA5 ISA 8.3/8.4 (count=1, type=2 "image", data_size=2B).
// ---------------------------------------------------------------------------
__device__ __forceinline__ void tdm_load_2d(unsigned lds_off, const void* gptr,
                                            unsigned tensor_d0, unsigned tensor_d1,
                                            unsigned tile_d0, unsigned tile_d1,
                                            unsigned long stride0) {
    unsigned long ga = (unsigned long)gptr;
    u32x4 g0;
    g0[0] = 1u;                                              // count=1, user desc
    g0[1] = lds_off;                                         // lds_addr (bytes)
    g0[2] = (unsigned)ga;                                    // global_addr[31:0]
    g0[3] = ((unsigned)(ga >> 32) & 0x1FFFFFFu) | 0x80000000u; // [56:32] | type=2
    u32x8 g1;
    g1[0] = (1u << 16) | (1u << 20) | (3u << 22) | (3u << 25); // 2B, pad 4dw/16dw
    g1[1] = (tensor_d0 & 0xFFFFu) << 16;                     // tensor_dim0[15:0]
    g1[2] = (tensor_d0 >> 16) | ((tensor_d1 & 0xFFFFu) << 16);
    g1[3] = (tensor_d1 >> 16) | (tile_d0 << 16);
    g1[4] = tile_d1;                                         // tile_dim2 = 0
    g1[5] = (unsigned)stride0;                               // dim0 stride (elems)
    g1[6] = (unsigned)(stride0 >> 32) & 0xFFFFu;             // stride[47:32]
    g1[7] = 0u;
    asm volatile("tensor_load_to_lds %0, %1" :: "s"(g0), "s"(g1) : "memory");
}

// A-fragment per ISA 7.12.2: lane row = lane&15, K halves at
// [kb..kb+7] and [kb+16..kb+23], kb = (lane>>4)*8.
__device__ __forceinline__ bf16x16 load_frag_a(const __bf16* rowp, int lane) {
    const int kb = (lane >> 4) * 8;
    bf16x8 lo = *(const bf16x8*)(rowp + kb);
    bf16x8 hi = *(const bf16x8*)(rowp + kb + 16);
    bf16x16 r;
#pragma unroll
    for (int i = 0; i < 8; ++i) { r[i] = lo[i]; r[i + 8] = hi[i]; }
    return r;
}
// B-fragment: lane col = lane&15, K halves at [kb..kb+7] and [kb+16..kb+23], kb = (lane>>4)*8.
__device__ __forceinline__ bf16x16 load_frag_b(const __bf16* rowp, int lane) {
    const int kb = (lane >> 4) * 8;
    bf16x8 lo = *(const bf16x8*)(rowp + kb);
    bf16x8 hi = *(const bf16x8*)(rowp + kb + 16);
    bf16x16 r;
#pragma unroll
    for (int i = 0; i < 8; ++i) { r[i] = lo[i]; r[i + 8] = hi[i]; }
    return r;
}

// ---------------------------------------------------------------------------
// Batched strided bf16 GEMM: C = alpha * A(MxK) * op(B) + bias, optional ReLU.
// TRANS_B=true : B is NxK row-major (Linear weight, C = A * B^T), TDM-staged.
// TRANS_B=false: B is KxN row-major (C = A * B), manually staged (transpose).
// A tiles always staged by TDM. LDS double-buffered: TDM of tile kt+1
// overlaps WMMA of tile kt. 256 threads = 8 waves; block tile 128x128,
// K-tile 32 (one WMMA step); wave tile 64x32 (4x2 accumulators).
// ---------------------------------------------------------------------------
template <bool TRANS_B, bool RELU>
__global__ __launch_bounds__(256)
void gemm_wmma_bf16(const __bf16* __restrict__ A, const __bf16* __restrict__ B,
                    const float* __restrict__ bias, __bf16* __restrict__ C,
                    int M, int N, int K, int lda, int ldb, int ldc,
                    int batchInner,
                    long sAo, long sAi, long sBo, long sBi, long sCo, long sCi,
                    float alpha) {
    constexpr int BM = 128, BN = 128, BK = 32, LDK = 40;  // 80B rows, conflict-free
    __shared__ __bf16 As[2][BM][LDK] __attribute__((aligned(16)));
    __shared__ __bf16 Bs[2][BN][LDK] __attribute__((aligned(16)));

    const int z = blockIdx.z;
    const int bo = z / batchInner, bi = z % batchInner;
    A += bo * sAo + bi * sAi;
    B += bo * sBo + bi * sBi;
    C += bo * sCo + bi * sCi;

    const int tid  = threadIdx.x;
    const int lane = tid & 31;
    const int wave = tid >> 5;
    const int wx   = wave & 3;   // N direction: 4 waves * 32 cols
    const int wy   = wave >> 2;  // M direction: 2 waves * 64 rows
    const int m0   = blockIdx.y * BM;
    const int n0   = blockIdx.x * BN;
    const int lrow = lane & 15;

    f32x8 acc[4][2];
#pragma unroll
    for (int mi = 0; mi < 4; ++mi)
#pragma unroll
        for (int ni = 0; ni < 2; ++ni)
#pragma unroll
            for (int i = 0; i < 8; ++i) acc[mi][ni][i] = 0.0f;

    const int ktiles = (K + BK - 1) / BK;

    auto stage_tdm = [&](int kt, int buf) {
        const int kk = kt * BK;
        tdm_load_2d((unsigned)(size_t)(void*)&As[buf][0][0],
                    (const char*)A + ((long)m0 * lda + kk) * 2,
                    (unsigned)(K - kk), (unsigned)(M - m0), BK, BM,
                    (unsigned long)lda);
        if (TRANS_B)
            tdm_load_2d((unsigned)(size_t)(void*)&Bs[buf][0][0],
                        (const char*)B + ((long)n0 * ldb + kk) * 2,
                        (unsigned)(K - kk), (unsigned)(N - n0), BK, BN,
                        (unsigned long)ldb);
    };
    auto stage_b_manual = [&](int kt, int buf) {
        const int kk = kt * BK;
#pragma unroll
        for (int it = 0; it < 16; ++it) {
            int idx = tid + it * 256;              // 4096 elems, coalesced over n
            int n = idx & 127, k = idx >> 7;
            int gn = n0 + n, gk = kk + k;
            __bf16 v = bf_zero();
            if (gn < N && gk < K) v = B[(long)gk * ldb + gn];
            Bs[buf][n][k] = v;
        }
    };

    // prologue: tile 0 into buffer 0
    if (wave == 0) stage_tdm(0, 0);
    if (!TRANS_B) stage_b_manual(0, 0);
    if (wave == 0) wait_tensorcnt0();
    __syncthreads();

    for (int kt = 0; kt < ktiles; ++kt) {
        const int cur = kt & 1, nbuf = cur ^ 1;
        if (kt + 1 < ktiles) {                     // overlap DMA with WMMA
            if (wave == 0) stage_tdm(kt + 1, nbuf);
            if (!TRANS_B) stage_b_manual(kt + 1, nbuf);
        }

        bf16x16 af[4], bfr[2];
#pragma unroll
        for (int mi = 0; mi < 4; ++mi)
            af[mi] = load_frag_a(&As[cur][wy * 64 + mi * 16 + lrow][0], lane);
#pragma unroll
        for (int ni = 0; ni < 2; ++ni)
            bfr[ni] = load_frag_b(&Bs[cur][wx * 32 + ni * 16 + lrow][0], lane);
#pragma unroll
        for (int mi = 0; mi < 4; ++mi)
#pragma unroll
            for (int ni = 0; ni < 2; ++ni)
                acc[mi][ni] = __builtin_amdgcn_wmma_f32_16x16x32_bf16(
                    false, af[mi], false, bfr[ni], (short)0, acc[mi][ni],
                    false, false);
        __syncthreads();                // everyone done reading cur
        if (wave == 0) wait_tensorcnt0();
        __syncthreads();                // next buffer fully landed in LDS
    }

    // epilogue: C/D layout -> row = i + 8*(lane>>4), col = lane&15
    const int rhalf = (lane >> 4) * 8;
#pragma unroll
    for (int mi = 0; mi < 4; ++mi) {
#pragma unroll
        for (int ni = 0; ni < 2; ++ni) {
            const int col = n0 + wx * 32 + ni * 16 + lrow;
            if (col >= N) continue;
            const float bv = bias ? bias[col] : 0.f;
#pragma unroll
            for (int i = 0; i < 8; ++i) {
                const int row = m0 + wy * 64 + mi * 16 + rhalf + i;
                if (row < M) {
                    float v = acc[mi][ni][i] * alpha + bv;
                    if (RELU) v = v > 0.f ? v : 0.f;
                    C[(long)row * ldc + col] = f2bf(v);
                }
            }
        }
    }
}

// ---------------------------------------------------------------------------
// fp32 -> bf16 bulk convert (weights / x), 4 elems per thread.
// ---------------------------------------------------------------------------
__global__ __launch_bounds__(256)
void cvt_kernel(const float* __restrict__ in, __bf16* __restrict__ out, long n) {
    long i = (long)blockIdx.x * 1024 + (long)threadIdx.x * 4;
    if (i + 3 < n) {
        float4 v = *(const float4*)(in + i);
        out[i + 0] = f2bf(v.x); out[i + 1] = f2bf(v.y);
        out[i + 2] = f2bf(v.z); out[i + 3] = f2bf(v.w);
    } else {
        for (long j = i; j < n; ++j) out[j] = f2bf(in[j]);
    }
}

// ---------------------------------------------------------------------------
// Row softmax on bf16 scores (in-place), optional relative-position bias.
// One block per row; rows = batch*H*Lrows; cols <= 1024.  fp32 internally.
// ---------------------------------------------------------------------------
__global__ __launch_bounds__(256)
void softmax_kernel(__bf16* __restrict__ scores, const float* __restrict__ rel_bias,
                    int cols, int Lrows, int H) {
    const long row = blockIdx.x;
    __bf16* p = scores + row * (long)cols;
    const int tid = threadIdx.x;
    const int l = (int)(row % Lrows);
    const int h = (int)((row / Lrows) % H);

    float vals[4];
    float mx = -3.0e38f;
#pragma unroll
    for (int it = 0; it < 4; ++it) {
        int c = tid + it * 256;
        float v = -3.0e38f;
        if (c < cols) {
            v = bf2f(p[c]);
            if (rel_bias) v += rel_bias[(long)(c - l + 1023) * H + h];
        }
        vals[it] = v;
        mx = fmaxf(mx, v);
    }
    __shared__ float red[256];
    red[tid] = mx; __syncthreads();
    for (int s = 128; s > 0; s >>= 1) {
        if (tid < s) red[tid] = fmaxf(red[tid], red[tid + s]);
        __syncthreads();
    }
    mx = red[0]; __syncthreads();

    float sum = 0.f;
#pragma unroll
    for (int it = 0; it < 4; ++it) {
        float e = (vals[it] > -1.0e37f) ? __expf(vals[it] - mx) : 0.f;
        vals[it] = e; sum += e;
    }
    red[tid] = sum; __syncthreads();
    for (int s = 128; s > 0; s >>= 1) {
        if (tid < s) red[tid] += red[tid + s];
        __syncthreads();
    }
    const float inv = 1.f / red[0];
#pragma unroll
    for (int it = 0; it < 4; ++it) {
        int c = tid + it * 256;
        if (c < cols) p[c] = f2bf(vals[it] * inv);
    }
}

// ---------------------------------------------------------------------------
// out[row] = LayerNorm(a + b (+ c)) * gamma + beta.  D = 768 (3 per thread).
// bf16 inputs, fp32 math; OutT = __bf16 (fusion halves) or float (d_out).
// ---------------------------------------------------------------------------
__device__ __forceinline__ void store_val(float* p, float v)  { *p = v; }
__device__ __forceinline__ void store_val(__bf16* p, float v) { *p = f2bf(v); }

template <typename OutT>
__global__ __launch_bounds__(256)
void ln_residual_kernel(const __bf16* __restrict__ a, int lda,
                        const __bf16* __restrict__ b, int ldb,
                        const __bf16* __restrict__ c, int ldc,
                        const float* __restrict__ gamma,
                        const float* __restrict__ beta,
                        OutT* __restrict__ out, int ldo, int D) {
    const long row = blockIdx.x;
    const __bf16* pa = a + row * (long)lda;
    const __bf16* pb = b ? b + row * (long)ldb : nullptr;
    const __bf16* pc = c ? c + row * (long)ldc : nullptr;
    const int tid = threadIdx.x;

    float vals[3];
    float s = 0.f;
#pragma unroll
    for (int it = 0; it < 3; ++it) {
        int d = tid + it * 256;
        float v = 0.f;
        if (d < D) {
            v = bf2f(pa[d]);
            if (pb) v += bf2f(pb[d]);
            if (pc) v += bf2f(pc[d]);
        }
        vals[it] = v; s += v;
    }
    __shared__ float red[256];
    red[tid] = s; __syncthreads();
    for (int st = 128; st > 0; st >>= 1) {
        if (tid < st) red[tid] += red[tid + st];
        __syncthreads();
    }
    const float mu = red[0] / (float)D; __syncthreads();

    float sq = 0.f;
#pragma unroll
    for (int it = 0; it < 3; ++it) {
        int d = tid + it * 256;
        if (d < D) { float dl = vals[it] - mu; sq += dl * dl; }
    }
    red[tid] = sq; __syncthreads();
    for (int st = 128; st > 0; st >>= 1) {
        if (tid < st) red[tid] += red[tid + st];
        __syncthreads();
    }
    const float rstd = rsqrtf(red[0] / (float)D + 1e-5f);
    OutT* po = out + row * (long)ldo;
#pragma unroll
    for (int it = 0; it < 3; ++it) {
        int d = tid + it * 256;
        if (d < D) store_val(po + d, (vals[it] - mu) * rstd * gamma[d] + beta[d]);
    }
}

// ---------------------------------------------------------------------------
// Batched bf16 2D transpose: in (Bz, R, C) -> out (Bz, C, R).  32x32 tiles.
// ---------------------------------------------------------------------------
__global__ __launch_bounds__(256)
void transpose_kernel(const __bf16* __restrict__ in, __bf16* __restrict__ out,
                      int R, int C) {
    __shared__ __bf16 tile[32][33];
    const long bz = blockIdx.z;
    in  += bz * (long)R * C;
    out += bz * (long)R * C;
    const int r0 = blockIdx.y * 32, c0 = blockIdx.x * 32;
    const int tx = threadIdx.x, ty = threadIdx.y;  // 32 x 8
#pragma unroll
    for (int i = ty; i < 32; i += 8) {
        int r = r0 + i, cc = c0 + tx;
        tile[i][tx] = (r < R && cc < C) ? in[(long)r * C + cc] : bf_zero();
    }
    __syncthreads();
#pragma unroll
    for (int i = ty; i < 32; i += 8) {
        int cc = c0 + i, r = r0 + tx;
        if (cc < C && r < R) out[(long)cc * R + r] = tile[tx][i];
    }
}

// ---------------------------------------------------------------------------
extern "C" void kernel_launch(void* const* d_in, const int* in_sizes, int n_in,
                              void* d_out, int out_size, void* d_ws, size_t ws_size,
                              hipStream_t stream) {
    (void)in_sizes; (void)n_in; (void)out_size; (void)ws_size;

    const float* x        = (const float*)d_in[0];
    const float* wq       = (const float*)d_in[1];
    const float* bq       = (const float*)d_in[2];
    const float* wk       = (const float*)d_in[3];
    const float* bk       = (const float*)d_in[4];
    const float* wv       = (const float*)d_in[5];
    const float* bv       = (const float*)d_in[6];
    const float* wo       = (const float*)d_in[7];
    const float* bo       = (const float*)d_in[8];
    const float* rel_bias = (const float*)d_in[9];
    const float* cin_w    = (const float*)d_in[10];
    const float* cin_b    = (const float*)d_in[11];
    const float* cout_w   = (const float*)d_in[12];
    const float* cout_b   = (const float*)d_in[13];
    const float* fw1      = (const float*)d_in[14];
    const float* fb1      = (const float*)d_in[15];
    const float* fw2      = (const float*)d_in[16];
    const float* fb2      = (const float*)d_in[17];
    const float* g_seq    = (const float*)d_in[18];
    const float* b_seq    = (const float*)d_in[19];
    const float* g_chan   = (const float*)d_in[20];
    const float* b_chan   = (const float*)d_in[21];
    const float* g_ffn    = (const float*)d_in[22];
    const float* b_ffn    = (const float*)d_in[23];
    float* out = (float*)d_out;

    const long Bn = 4, L = 1024, D = 768, H = 16;
    const long BL = Bn * L;                 // 4096
    const long LD = L * D;                  // 786432

    __bf16* wsb = (__bf16*)d_ws;
    size_t off = 0;
    auto nxt = [&](size_t n) {
        __bf16* p = wsb + off;
        off += (n + 63) & ~(size_t)63;
        return p;
    };
    // bf16 copies of fp32 inputs (converted once per launch)
    __bf16* xb     = nxt(BL * D);
    __bf16* wqb    = nxt(768 * 768);
    __bf16* wkb    = nxt(768 * 768);
    __bf16* wvb    = nxt(768 * 768);
    __bf16* wob    = nxt(768 * 768);
    __bf16* cinwb  = nxt(3072 * 1024);
    __bf16* coutwb = nxt(1024 * 1024);
    __bf16* fw1b   = nxt(1536 * 1536);
    __bf16* fw2b   = nxt(768 * 1536);
    // bf16 intermediates
    __bf16* q      = nxt(BL * D);
    __bf16* kbuf   = nxt(BL * D);
    __bf16* vbuf   = nxt(BL * D);
    __bf16* attno  = nxt(BL * D);
    __bf16* sc     = nxt((size_t)Bn * H * L * L);   // 128 MB, reused by chan
    __bf16* xT     = nxt((size_t)Bn * D * L);
    __bf16* qkvc   = nxt((size_t)Bn * D * 3 * L);
    __bf16* chant  = nxt((size_t)Bn * D * L);
    __bf16* chanoT = nxt((size_t)Bn * D * L);
    __bf16* chanat = nxt(BL * D);
    __bf16* fusion = nxt(BL * 2 * D);
    __bf16* h1     = nxt(BL * 2 * D);
    __bf16* ffnout = nxt(BL * D);

    const dim3 blk(256);
    auto grd = [](long M, long N, long Z) {
        return dim3((unsigned)((N + 127) / 128), (unsigned)((M + 127) / 128),
                    (unsigned)Z);
    };
    auto cvt = [&](const float* src, __bf16* dst, long n) {
        cvt_kernel<<<(unsigned)((n + 1023) / 1024), blk, 0, stream>>>(src, dst, n);
    };

    // ---- one-time fp32 -> bf16 conversions ----
    cvt(x, xb, BL * D);
    cvt(wq, wqb, 768 * 768);     cvt(wk, wkb, 768 * 768);
    cvt(wv, wvb, 768 * 768);     cvt(wo, wob, 768 * 768);
    cvt(cin_w, cinwb, 3072 * 1024);
    cvt(cout_w, coutwb, 1024 * 1024);
    cvt(fw1, fw1b, 1536 * 1536); cvt(fw2, fw2b, 768 * 1536);

    // ---- sequence branch: Q/K/V projections (Linear: x @ W^T + b) ----
    gemm_wmma_bf16<true, false><<<grd(4096, 768, 1), blk, 0, stream>>>(
        xb, wqb, bq, q, 4096, 768, 768, 768, 768, 768, 1, 0, 0, 0, 0, 0, 0, 1.0f);
    gemm_wmma_bf16<true, false><<<grd(4096, 768, 1), blk, 0, stream>>>(
        xb, wkb, bk, kbuf, 4096, 768, 768, 768, 768, 768, 1, 0, 0, 0, 0, 0, 0, 1.0f);
    gemm_wmma_bf16<true, false><<<grd(4096, 768, 1), blk, 0, stream>>>(
        xb, wvb, bv, vbuf, 4096, 768, 768, 768, 768, 768, 1, 0, 0, 0, 0, 0, 0, 1.0f);

    // scores[b,h] = (Q_bh @ K_bh^T) / sqrt(48)   (64 batched 1024x1024x48)
    gemm_wmma_bf16<true, false><<<grd(1024, 1024, 64), blk, 0, stream>>>(
        q, kbuf, nullptr, sc, 1024, 1024, 48, 768, 768, 1024, 16,
        LD, 48, LD, 48, (long)H * L * L, L * L, 0.14433756729740643f);

    softmax_kernel<<<(unsigned)(Bn * H * L), blk, 0, stream>>>(
        sc, rel_bias, 1024, 1024, 16);

    // attn @ V  (64 batched 1024x48x1024, B is KxN row-major)
    gemm_wmma_bf16<false, false><<<grd(1024, 48, 64), blk, 0, stream>>>(
        sc, vbuf, nullptr, attno, 1024, 48, 1024, 1024, 768, 768, 16,
        (long)H * L * L, L * L, LD, 48, LD, 48, 1.0f);

    // output projection -> seq_attn (reuse q buffer)
    gemm_wmma_bf16<true, false><<<grd(4096, 768, 1), blk, 0, stream>>>(
        attno, wob, bo, q, 4096, 768, 768, 768, 768, 768, 1, 0, 0, 0, 0, 0, 0, 1.0f);

    // x_seq = LN(x + seq_attn) -> fusion[:, 0:768]
    ln_residual_kernel<<<(unsigned)BL, blk, 0, stream>>>(
        xb, 768, q, 768, (const __bf16*)nullptr, 0, g_seq, b_seq,
        fusion, 1536, 768);

    // ---- channel branch ----
    transpose_kernel<<<dim3(24, 32, 4), dim3(32, 8), 0, stream>>>(xb, xT, 1024, 768);

    // qkv = xT @ cin_w^T + cin_b   (per batch: 768 x 3072 x 1024)
    gemm_wmma_bf16<true, false><<<grd(768, 3072, 4), blk, 0, stream>>>(
        xT, cinwb, cin_b, qkvc, 768, 3072, 1024, 1024, 1024, 3072, 1,
        (long)D * L, 0, 0, 0, (long)D * 3 * L, 0, 1.0f);

    // chan scores[b,h] = (q_bh @ k_bh^T) / 8   (64 batched 768x768x64)
    gemm_wmma_bf16<true, false><<<grd(768, 768, 64), blk, 0, stream>>>(
        qkvc, qkvc + 1024, nullptr, sc, 768, 768, 64, 3072, 3072, 768, 16,
        (long)D * 3 * L, 64, (long)D * 3 * L, 64, (long)H * D * D, (long)D * D,
        0.125f);

    softmax_kernel<<<(unsigned)(Bn * H * D), blk, 0, stream>>>(
        sc, (const float*)nullptr, 768, 768, 16);

    // attn @ v  (64 batched 768x64x768)
    gemm_wmma_bf16<false, false><<<grd(768, 64, 64), blk, 0, stream>>>(
        sc, qkvc + 2048, nullptr, chant, 768, 64, 768, 768, 3072, 1024, 16,
        (long)H * D * D, (long)D * D, (long)D * 3 * L, 64, (long)D * L, 64, 1.0f);

    // channel output projection (per batch: 768 x 1024 x 1024)
    gemm_wmma_bf16<true, false><<<grd(768, 1024, 4), blk, 0, stream>>>(
        chant, coutwb, cout_b, chanoT, 768, 1024, 1024, 1024, 1024, 1024, 1,
        (long)D * L, 0, 0, 0, (long)D * L, 0, 1.0f);

    transpose_kernel<<<dim3(32, 24, 4), dim3(32, 8), 0, stream>>>(
        chanoT, chanat, 768, 1024);

    // x_chan = LN(x + chan_attn) -> fusion[:, 768:1536]
    ln_residual_kernel<<<(unsigned)BL, blk, 0, stream>>>(
        xb, 768, chanat, 768, (const __bf16*)nullptr, 0, g_chan, b_chan,
        fusion + 768, 1536, 768);

    // ---- FFN on fused buffer ----
    gemm_wmma_bf16<true, true><<<grd(4096, 1536, 1), blk, 0, stream>>>(
        fusion, fw1b, fb1, h1, 4096, 1536, 1536, 1536, 1536, 1536, 1,
        0, 0, 0, 0, 0, 0, 1.0f);
    gemm_wmma_bf16<true, false><<<grd(4096, 768, 1), blk, 0, stream>>>(
        h1, fw2b, fb2, ffnout, 4096, 768, 1536, 1536, 1536, 768, 1,
        0, 0, 0, 0, 0, 0, 1.0f);

    // out = LN(x_seq + x_chan + ffn_out)
    ln_residual_kernel<<<(unsigned)BL, blk, 0, stream>>>(
        fusion, 1536, fusion + 768, 1536, ffnout, 768, g_ffn, b_ffn,
        out, 768, 768);
}
